// GraphAttentionLayer_54262616818210
// MI455X (gfx1250) — matrix operand order repaired and checked by reading, not compiled, not run on hardware
//
#include <hip/hip_runtime.h>
#include <hip/hip_bf16.h>

// ---------------------------------------------------------------------------
// GraphAttentionLayer for MI455X (gfx1250, wave32, WMMA).
//
//   K1: transpose+split W (16384x256 fp32) -> Wt_hi/Wt_lo (256x16384 bf16)
//   K2: Wh = hf @ W via V_WMMA_F32_16X16X32_BF16; fp32 emulated as 3 bf16
//       WMMAs (hi*hi + hi*lo + lo*hi). Two n's packed per 16-row M tile.
//       A panel converted to packed bf16 fragments ONCE per block in LDS.
//   K3: attention + softmax + graph-norm + att@Wh + ELU (tiny)
//
// Workspace (~18.6 MB): Wt_hi[8MB] | Wt_lo[8MB] | Wh[1.75MB]
// ---------------------------------------------------------------------------

typedef __attribute__((ext_vector_type(8)))  float  v8f;
typedef __attribute__((ext_vector_type(16))) __bf16 v16bf;

union BfVec {
    v16bf          v;
    uint4          q[2];   // two 16B halves (16B-aligned LDS access)
    unsigned       u[8];
    unsigned short s[16];
};

#define CT_DIM   16384   // C*T
#define V_DIM    7
#define O_DIM    256
#define N_DIM    256
#define KT       256     // K tile staged in LDS
#define FRAG_STRIDE 24   // ushorts per packed A fragment record (48B: 16B-aligned,
                         // stride-12-dword -> only 2-way LDS bank overlap)

// ---------------------------------------------------------------------------
// K1: W[k][o] fp32 -> Wt_hi/Wt_lo[o][k] bf16 (hi = trunc bf16, lo = bf16(x-hi))
// ---------------------------------------------------------------------------
__global__ void __launch_bounds__(256)
k_transpose_split_w(const float* __restrict__ W,
                    unsigned short* __restrict__ wtH,
                    unsigned short* __restrict__ wtL) {
    __shared__ float tile[32][33];
    const int kb = blockIdx.x * 32;
    const int ob = blockIdx.y * 32;
    const int tx = threadIdx.x & 31;
    const int ty = threadIdx.x >> 5;

#pragma unroll
    for (int i = 0; i < 4; ++i) {
        const int row = ty + i * 8;
        tile[row][tx] = W[(size_t)(kb + row) * O_DIM + ob + tx];
    }
    __syncthreads();
#pragma unroll
    for (int i = 0; i < 4; ++i) {
        const int orow = ty + i * 8;
        const float f  = tile[tx][orow];      // = W[kb+tx][ob+orow]
        const unsigned hu = __float_as_uint(f) & 0xFFFF0000u;
        const float    lf = f - __uint_as_float(hu);
        const size_t dst  = (size_t)(ob + orow) * CT_DIM + kb + tx;
        wtH[dst] = (unsigned short)(hu >> 16);
        wtL[dst] = (unsigned short)(__float_as_uint(lf) >> 16);
    }
}

// ---------------------------------------------------------------------------
// K2: Wh = hf @ W.  Grid (128 n-pairs, 2 o-halves) x 256 threads (8 wave32).
// A tile rows: 0..6 -> n0, 7..13 -> n1, 14..15 zero.  Wave w owns o-tile
// (blockIdx.y*8 + w).  Per KT tile:
//   phase1: cooperative coalesced load of h panel -> ldsF[k][16] (fp32)
//   phase2: 256 threads convert 256 fragments (8 chunks x 32 lanes) to packed
//           bf16 hi/lo in per-lane A-fragment order -> ldsAH / ldsAL
//   phase3: WMMA loop: 2x16B LDS A loads + 1x32B global B load per fragment.
// ---------------------------------------------------------------------------
__global__ void __launch_bounds__(256)
k_gat_gemm_wmma(const float* __restrict__ h,
                const unsigned short* __restrict__ wtH,
                const unsigned short* __restrict__ wtL,
                float* __restrict__ Wh) {
    __shared__ float          ldsF[KT * 16];                    // 16 KB
    __shared__ unsigned short ldsAH[(KT / 32) * 32 * FRAG_STRIDE]; // 12 KB
    __shared__ unsigned short ldsAL[(KT / 32) * 32 * FRAG_STRIDE]; // 12 KB

    const int t    = threadIdx.x;
    const int wave = t >> 5;
    const int lane = t & 31;
    const int l    = lane & 15;
    const int hfl  = lane >> 4;

    const int n0 = blockIdx.x * 2;
    const int n1 = n0 + 1;
    const float* H0 = h + (size_t)n0 * CT_DIM * V_DIM;
    const float* H1 = h + (size_t)n1 * CT_DIM * V_DIM;

    // Pad rows 14,15 once; loads only ever write rows 0..13.
    ldsF[t * 16 + 14] = 0.0f;
    ldsF[t * 16 + 15] = 0.0f;

    const int ob = (blockIdx.y * 8 + wave) * 16;
    const unsigned short* pBH = wtH + (size_t)(ob + l) * CT_DIM + hfl * 16;
    const unsigned short* pBL = wtL + (size_t)(ob + l) * CT_DIM + hfl * 16;

    v8f acc = {0.f, 0.f, 0.f, 0.f, 0.f, 0.f, 0.f, 0.f};

    for (int kt = 0; kt < CT_DIM; kt += KT) {
        __syncthreads();
        // ---- phase1: stage fp32 A panel (coalesced; scatter k*7+m -> [k][m])
#pragma unroll
        for (int p = 0; p < 7; ++p) {
            const int idx = p * 256 + t;          // 0..1791 = k*7+m
            const int k   = idx / 7;
            const int m   = idx - k * 7;
            ldsF[k * 16 + m]     = H0[(size_t)kt * 7 + idx];
            ldsF[k * 16 + 7 + m] = H1[(size_t)kt * 7 + idx];
        }
        __syncthreads();

        // ---- phase2: thread t converts fragment (chunk = t>>5, lane = t&31)
        {
            BfVec hi, lo;
#pragma unroll
            for (int e = 0; e < 16; ++e) {
                const int j   = e >> 1;
                const int pos = e & 1;
                // 16-bit A 16x32: contiguous k = 16*hfl + e (matches B)
                const int kl  = hfl * 16 + e;
                const float f = ldsF[((wave << 5) + kl) * 16 + l];
                const unsigned hu = __float_as_uint(f) & 0xFFFF0000u;
                const float    lf = f - __uint_as_float(hu);
                hi.s[e] = (unsigned short)(hu >> 16);
                lo.s[e] = (unsigned short)(__float_as_uint(lf) >> 16);
            }
            unsigned short* dh = &ldsAH[t * FRAG_STRIDE];
            unsigned short* dl = &ldsAL[t * FRAG_STRIDE];
            *(uint4*)(dh)     = hi.q[0];
            *(uint4*)(dh + 8) = hi.q[1];
            *(uint4*)(dl)     = lo.q[0];
            *(uint4*)(dl + 8) = lo.q[1];
        }
        __syncthreads();

        // ---- phase3: pure load + WMMA
#pragma unroll
        for (int c = 0; c < KT / 32; ++c) {
            BfVec Ahi, Alo;
            const unsigned short* ah = &ldsAH[(c * 32 + lane) * FRAG_STRIDE];
            const unsigned short* al = &ldsAL[(c * 32 + lane) * FRAG_STRIDE];
            Ahi.q[0] = *(const uint4*)(ah);
            Ahi.q[1] = *(const uint4*)(ah + 8);
            Alo.q[0] = *(const uint4*)(al);
            Alo.q[1] = *(const uint4*)(al + 8);

            // B fragment: 16 K-contiguous bf16 -> one aligned 32B load
            const v16bf Bh = *(const v16bf*)(pBH + kt + c * 32);
            const v16bf Bl = *(const v16bf*)(pBL + kt + c * 32);

            acc = __builtin_amdgcn_wmma_f32_16x16x32_bf16(false, Ahi.v, false, Bh, (short)0, acc, false, false);
            acc = __builtin_amdgcn_wmma_f32_16x16x32_bf16(false, Ahi.v, false, Bl, (short)0, acc, false, false);
            acc = __builtin_amdgcn_wmma_f32_16x16x32_bf16(false, Alo.v, false, Bh, (short)0, acc, false, false);
        }
    }

    // C/D layout: VGPR r, lanes 0-15 -> M=r, lanes 16-31 -> M=8+r, N=l.
#pragma unroll
    for (int r = 0; r < 8; ++r) {
        const int M = r + 8 * hfl;
        if (M < 14) {
            const int n = (M < 7) ? n0 : n1;
            const int v = (M < 7) ? M : M - 7;
            Wh[((size_t)n * V_DIM + v) * O_DIM + ob + l] = acc[r];
        }
    }
}

// ---------------------------------------------------------------------------
// K3: per-n attention + graph normalization + att@Wh + ELU.  One block per n.
// ---------------------------------------------------------------------------
__global__ void __launch_bounds__(256)
k_gat_attention(const float* __restrict__ Wh,
                const float* __restrict__ a,
                const float* __restrict__ Bp,
                float* __restrict__ out) {
    __shared__ float sWh[V_DIM * O_DIM];
    __shared__ float s1[V_DIM], s2[V_DIM];
    __shared__ float att[49], att2[49], adjn[49];

    const int n = blockIdx.x;
    const int t = threadIdx.x;

#pragma unroll
    for (int v = 0; v < V_DIM; ++v)
        sWh[v * O_DIM + t] = Wh[((size_t)n * V_DIM + v) * O_DIM + t];
    __syncthreads();

    const int w    = t >> 5;
    const int lane = t & 31;
    if (w < 7) {
        float p1 = 0.f, p2 = 0.f;
        for (int o = lane; o < O_DIM; o += 32) {
            const float x = sWh[w * O_DIM + o];
            p1 += x * a[o];
            p2 += x * a[O_DIM + o];
        }
#pragma unroll
        for (int off = 16; off > 0; off >>= 1) {
            p1 += __shfl_down(p1, off, 32);
            p2 += __shfl_down(p2, off, 32);
        }
        if (lane == 0) { s1[w] = p1; s2[w] = p2; }
    } else if (t == 224) {
        float adj[49];
        float mn = 1e30f, mx = -1e30f;
        for (int i = 0; i < 7; ++i)
            for (int j = 0; j < 7; ++j) {
                const float x = Bp[i * 7 + j] + 1e-6f + (i == j ? 1.0f : 0.0f);
                adj[i * 7 + j] = x;
                mn = fminf(mn, x);
                mx = fmaxf(mx, x);
            }
        const float inv = 1.0f / (mx - mn);
        float d12[7];
        for (int i = 0; i < 7; ++i) {
            float s = 0.f;
            for (int j = 0; j < 7; ++j) {
                adj[i * 7 + j] = (adj[i * 7 + j] - mn) * inv;
                s += adj[i * 7 + j];
            }
            d12[i] = rsqrtf(s);
        }
        for (int i = 0; i < 7; ++i)
            for (int j = 0; j < 7; ++j)
                adjn[i * 7 + j] = d12[i] * adj[i * 7 + j] * d12[j];
    }
    __syncthreads();

    if (t < 7) {  // leaky_relu + row softmax
        float e[7], mx = -1e30f;
#pragma unroll
        for (int j = 0; j < 7; ++j) {
            float x = s1[t] + s2[j];
            x = (x > 0.f) ? x : 0.2f * x;
            e[j] = x;
            mx = fmaxf(mx, x);
        }
        float s = 0.f;
#pragma unroll
        for (int j = 0; j < 7; ++j) { e[j] = __expf(e[j] - mx); s += e[j]; }
        const float inv = 1.0f / s;
#pragma unroll
        for (int j = 0; j < 7; ++j) att[t * 7 + j] = e[j] * inv;
    }
    __syncthreads();

    if (t < 49) {  // att2 = adjn @ att
        const int i = t / 7, j = t - (t / 7) * 7;
        float acc = 0.f;
#pragma unroll
        for (int k = 0; k < 7; ++k) acc += adjn[i * 7 + k] * att[k * 7 + j];
        att2[t] = acc;
    }
    __syncthreads();

#pragma unroll
    for (int v = 0; v < V_DIM; ++v) {  // h_prime = att2 @ Wh ; ELU
        float acc = 0.f;
#pragma unroll
        for (int j = 0; j < 7; ++j) acc += att2[v * 7 + j] * sWh[j * O_DIM + t];
        out[((size_t)n * V_DIM + v) * O_DIM + t] =
            (acc > 0.f) ? acc : (__expf(acc) - 1.0f);
    }
}

// ---------------------------------------------------------------------------
extern "C" void kernel_launch(void* const* d_in, const int* in_sizes, int n_in,
                              void* d_out, int out_size, void* d_ws, size_t ws_size,
                              hipStream_t stream) {
    (void)in_sizes; (void)n_in; (void)out_size; (void)ws_size;

    const float* h  = (const float*)d_in[0];   // (256,256,64,7)
    const float* W  = (const float*)d_in[1];   // (16384,256)
    const float* a  = (const float*)d_in[2];   // (512,1)
    const float* Bp = (const float*)d_in[3];   // (7,7)
    float* out = (float*)d_out;                // (256,7,256)

    char* ws = (char*)d_ws;
    unsigned short* wtH = (unsigned short*)ws;
    unsigned short* wtL = (unsigned short*)(ws + (size_t)O_DIM * CT_DIM * 2);
    float*          Wh  = (float*)(ws + (size_t)2 * O_DIM * CT_DIM * 2);

    k_transpose_split_w<<<dim3(CT_DIM / 32, O_DIM / 32), 256, 0, stream>>>(W, wtH, wtL);
    k_gat_gemm_wmma<<<dim3(N_DIM / 2, 2), 256, 0, stream>>>(h, wtH, wtL, Wh);
    k_gat_attention<<<dim3(N_DIM), 256, 0, stream>>>(Wh, a, Bp, out);
}
